// EmbeddingCellLSTM_17789754540431
// MI455X (gfx1250) — compile-verified, matrix operand order checked by reading, not run on hardware
//
#include <hip/hip_runtime.h>

// ---- problem constants ----
#define BB   32
#define SS   2048
#define IN_  64
#define EMB_ 32
#define HH   256
#define G4H  1024          // 4*H
#define KIN  96            // IN+EMB

typedef _Float16 v16h __attribute__((ext_vector_type(16)));
typedef _Float16 v8h  __attribute__((ext_vector_type(8)));
typedef float    v8f  __attribute__((ext_vector_type(8)));

union AFrag { v16h v; v8h h[2]; };

__device__ __forceinline__ float sigmoidf_dev(float x) {
    return 1.0f / (1.0f + __expf(-x));
}

// async global->LDS 16B per lane; IOFFSET applies to both LDS dst and global src
#define ASYNC_B128(ldsoff, gaddr, IMM)                                        \
    asm volatile("global_load_async_to_lds_b128 %0, %1, off offset:" #IMM     \
                 :: "v"(ldsoff), "v"(gaddr) : "memory")

// ---------------------------------------------------------------------------
// Pack [x | emb] -> xe_f16 [B*S, 96]
// ---------------------------------------------------------------------------
__global__ __launch_bounds__(256) void pack_xe_kernel(
    const float* __restrict__ x, const float* __restrict__ emb,
    _Float16* __restrict__ xe) {
    size_t idx = (size_t)blockIdx.x * 256 + threadIdx.x;
    if (idx >= (size_t)BB * SS * KIN) return;
    int    col = (int)(idx % KIN);
    size_t row = idx / KIN;                       // row = b*S + s
    float v = (col < IN_) ? x[row * IN_ + col]
                          : emb[row * EMB_ + (col - IN_)];
    xe[idx] = (_Float16)v;
}

// ---------------------------------------------------------------------------
// f32 -> f16 elementwise (weight conversion)
// ---------------------------------------------------------------------------
__global__ __launch_bounds__(256) void conv_f16_kernel(
    const float* __restrict__ src, _Float16* __restrict__ dst, int n) {
    int idx = blockIdx.x * 256 + threadIdx.x;
    if (idx < n) dst[idx] = (_Float16)src[idx];
}

// ---------------------------------------------------------------------------
// Input GEMM: zx[m, n] = xe[m, :] . W_ih[n, :] + b[n]
// m = b*S+s in [0,65536), n in [0,1024), K = 96
// B fragments for a k-step are loaded as a group (one clause) before the
// 4 WMMAs so load latency overlaps the previous k-step's WMMAs.
// ---------------------------------------------------------------------------
__global__ __launch_bounds__(256) void gemm_in_kernel(
    const _Float16* __restrict__ xe, const _Float16* __restrict__ wih,
    const float* __restrict__ bias, float* __restrict__ zx) {
    int lane  = threadIdx.x & 31;
    int gwave = blockIdx.x * 8 + (threadIdx.x >> 5);
    int mtile = gwave >> 4;            // 0..4095
    int ngrp  = gwave & 15;            // 0..15  -> 64 output columns
    int m0  = mtile * 16;
    int ln  = lane & 15;
    int hi8 = (lane >> 4) * 8;

    v8f acc[4];
    for (int gi = 0; gi < 4; ++gi) {
        float bv = bias[ngrp * 64 + gi * 16 + ln];
        for (int r = 0; r < 8; ++r) acc[gi][r] = bv;
    }

    const _Float16* arow = xe + (size_t)(m0 + ln) * KIN + hi8;
    const _Float16* brow[4];
#pragma unroll
    for (int gi = 0; gi < 4; ++gi)
        brow[gi] = wih + (size_t)(ngrp * 64 + gi * 16 + ln) * KIN + hi8;

#pragma unroll
    for (int kt = 0; kt < 3; ++kt) {
        AFrag a;
        a.h[0] = *(const v8h*)(arow + kt * 32);
        a.h[1] = *(const v8h*)(arow + kt * 32 + 16);
        AFrag bm[4];
#pragma unroll
        for (int gi = 0; gi < 4; ++gi) {
            bm[gi].h[0] = *(const v8h*)(brow[gi] + kt * 32);
            bm[gi].h[1] = *(const v8h*)(brow[gi] + kt * 32 + 16);
        }
#pragma unroll
        for (int gi = 0; gi < 4; ++gi)
            acc[gi] = __builtin_amdgcn_wmma_f32_16x16x32_f16(
                false, a.v, false, bm[gi].v, (short)0, acc[gi], false, false);
    }
#pragma unroll
    for (int gi = 0; gi < 4; ++gi) {
        int n = ngrp * 64 + gi * 16 + ln;
#pragma unroll
        for (int r = 0; r < 8; ++r) {
            int m = m0 + r + hi8;
            zx[(size_t)m * G4H + n] = acc[gi][r];
        }
    }
}

// ---------------------------------------------------------------------------
// Recurrent kernel: 2 independent workgroups (one 16-row batch tile each),
// 512 threads = 16 waves per WG. Wave t owns gate tiles {t,t+16,t+32,t+48}
// (i/f/g/o for hidden 16t..16t+15): gates + cell state stay in VGPRs.
// Next step's zx slice is double-buffered into LDS with async loads.
// ---------------------------------------------------------------------------
__global__ __launch_bounds__(512) void lstm_rec_kernel(
    const float* __restrict__ zx, const _Float16* __restrict__ whh,
    float* __restrict__ out) {
    __shared__ float    zbuf[2 * 16 * G4H];     // 128 KB: zx double buffer
    __shared__ _Float16 hbuf[16 * HH];          // 8 KB: h state as f16

    int tid  = threadIdx.x;
    int lane = tid & 31;
    int t    = tid >> 5;                        // wave index = hidden group
    int ln   = lane & 15;
    int hi8  = (lane >> 4) * 8;
    int hidx = t * 16 + ln;
    int B0   = blockIdx.x * 16;                 // batch tile base

    // zero h state
    {
        v8h z = {};
        *(v8h*)(&hbuf[tid * 8]) = z;
    }
    float cst[8];
#pragma unroll
    for (int r = 0; r < 8; ++r) cst[r] = 0.0f;

    const _Float16* bbase[4];
#pragma unroll
    for (int gi = 0; gi < 4; ++gi)
        bbase[gi] = whh + (size_t)(gi * 256 + hidx) * HH + hi8;
    const _Float16* abase = &hbuf[ln * HH + hi8];

    float* hout = out;                               // [B,S,H]
    float* hfin = out + (size_t)BB * SS * HH;        // [B,H]
    float* cfin = hfin + BB * HH;                    // [B,H]

    // async producer addressing: thread tid stages 32 contiguous floats of
    // row (tid>>5) at column (tid&31)*32 -> LDS byte offset tid*128
    int prow = tid >> 5;
    int pcol = (tid & 31) * 32;
    const float* gsrc0 = zx + ((size_t)(B0 + prow) * SS) * G4H + pcol;

    // prologue: stage zx for s = 0 into buffer 0
    {
        unsigned ldso = (unsigned)(size_t)(&zbuf[0]) + (unsigned)(tid * 128);
        unsigned long long ga = (unsigned long long)(size_t)gsrc0;
        ASYNC_B128(ldso, ga, 0);   ASYNC_B128(ldso, ga, 16);
        ASYNC_B128(ldso, ga, 32);  ASYNC_B128(ldso, ga, 48);
        ASYNC_B128(ldso, ga, 64);  ASYNC_B128(ldso, ga, 80);
        ASYNC_B128(ldso, ga, 96);  ASYNC_B128(ldso, ga, 112);
    }

    for (int s = 0; s < SS; ++s) {
        int cur = s & 1;
        // stage next step's zx into the other buffer while we compute
        if (s + 1 < SS) {
            unsigned ldso = (unsigned)(size_t)(&zbuf[(1 - cur) * 16 * G4H]) +
                            (unsigned)(tid * 128);
            unsigned long long ga =
                (unsigned long long)(size_t)(gsrc0 + (size_t)(s + 1) * G4H);
            ASYNC_B128(ldso, ga, 0);   ASYNC_B128(ldso, ga, 16);
            ASYNC_B128(ldso, ga, 32);  ASYNC_B128(ldso, ga, 48);
            ASYNC_B128(ldso, ga, 64);  ASYNC_B128(ldso, ga, 80);
            ASYNC_B128(ldso, ga, 96);  ASYNC_B128(ldso, ga, 112);
            asm volatile("s_wait_asynccnt 0x8" ::: "memory");
        } else {
            asm volatile("s_wait_asynccnt 0x0" ::: "memory");
        }
        __syncthreads();      // zbuf[cur] complete everywhere; hbuf visible

        // C init from staged input projection (LDS)
        const float* zc = &zbuf[cur * 16 * G4H];
        v8f acc[4];
#pragma unroll
        for (int gi = 0; gi < 4; ++gi) {
            int n = gi * 256 + hidx;
#pragma unroll
            for (int r = 0; r < 8; ++r)
                acc[gi][r] = zc[(r + hi8) * G4H + n];
        }
        // z += h @ W_hh^T  (K=256 as 8 steps of 16x16x32 f16 WMMA);
        // group the 8 B-fragment loads per k-step so they issue as one
        // clause and overlap the previous k-step's WMMAs.
#pragma unroll
        for (int kt = 0; kt < 8; ++kt) {
            AFrag a;
            a.h[0] = *(const v8h*)(abase + kt * 32);        // ds_load_b128
            a.h[1] = *(const v8h*)(abase + kt * 32 + 16);
            if (kt < 7)
                __builtin_prefetch(bbase[0] + (kt + 1) * 32, 0, 1);
            AFrag bm[4];
#pragma unroll
            for (int gi = 0; gi < 4; ++gi) {
                bm[gi].h[0] = *(const v8h*)(bbase[gi] + kt * 32);
                bm[gi].h[1] = *(const v8h*)(bbase[gi] + kt * 32 + 16);
            }
#pragma unroll
            for (int gi = 0; gi < 4; ++gi)
                acc[gi] = __builtin_amdgcn_wmma_f32_16x16x32_f16(
                    false, a.v, false, bm[gi].v, (short)0, acc[gi], false, false);
        }
        // gates + state update, all in registers
        float hv[8];
#pragma unroll
        for (int r = 0; r < 8; ++r) {
            float si = sigmoidf_dev(acc[0][r]);
            float sf = sigmoidf_dev(acc[1][r]);
            float tg = tanhf(acc[2][r]);
            float so = sigmoidf_dev(acc[3][r]);
            float c  = sf * cst[r] + si * tg;
            cst[r]   = c;
            float h  = so * tanhf(c);
            hv[r]    = h;
            int b = B0 + r + hi8;
            hout[((size_t)b * SS + s) * HH + hidx] = h;
            if (s == SS - 1) {
                hfin[b * HH + hidx] = h;
                cfin[b * HH + hidx] = c;
            }
        }
        __syncthreads();      // all hbuf reads of this step done
#pragma unroll
        for (int r = 0; r < 8; ++r)
            hbuf[(r + hi8) * HH + hidx] = (_Float16)hv[r];
    }
}

// ---------------------------------------------------------------------------
extern "C" void kernel_launch(void* const* d_in, const int* in_sizes, int n_in,
                              void* d_out, int out_size, void* d_ws, size_t ws_size,
                              hipStream_t stream) {
    const float* x    = (const float*)d_in[0];   // [B,S,IN]
    const float* emb  = (const float*)d_in[1];   // [B,S,EMB]
    const float* W_ih = (const float*)d_in[2];   // [4H, 96]
    const float* W_hh = (const float*)d_in[3];   // [4H, H]
    const float* bias = (const float*)d_in[4];   // [4H]
    float* out = (float*)d_out;

    // workspace layout
    char* ws = (char*)d_ws;
    float*    zx  = (float*)ws;                                   // 256 MB
    size_t off    = (size_t)BB * SS * G4H * sizeof(float);
    _Float16* xe  = (_Float16*)(ws + off);                        // 12.6 MB
    off          += (size_t)BB * SS * KIN * sizeof(_Float16);
    _Float16* wih = (_Float16*)(ws + off);                        // 192 KB
    off          += (size_t)G4H * KIN * sizeof(_Float16);
    _Float16* whh = (_Float16*)(ws + off);                        // 512 KB

    // 1) pack + convert inputs/weights to f16
    {
        size_t n = (size_t)BB * SS * KIN;
        pack_xe_kernel<<<(unsigned)((n + 255) / 256), 256, 0, stream>>>(x, emb, xe);
    }
    conv_f16_kernel<<<(G4H * KIN + 255) / 256, 256, 0, stream>>>(W_ih, wih, G4H * KIN);
    conv_f16_kernel<<<(G4H * HH  + 255) / 256, 256, 0, stream>>>(W_hh, whh, G4H * HH);

    // 2) bulk input GEMM (WMMA across full GPU): 65536x1024x96
    gemm_in_kernel<<<8192, 256, 0, stream>>>(xe, wih, bias, zx);

    // 3) sequential LSTM recurrence: 2 independent batch-tile workgroups
    lstm_rec_kernel<<<2, 512, 0, stream>>>(zx, whh, out);
}
